// LayoutImageBertGNN_54520314855555
// MI455X (gfx1250) — matrix operand order repaired and checked by reading, not compiled, not run on hardware
//
#include <hip/hip_runtime.h>
#include <math.h>

// ---------------------------------------------------------------------------
// Types for WMMA
// ---------------------------------------------------------------------------
typedef __attribute__((ext_vector_type(16))) __bf16 v16bf;
typedef __attribute__((ext_vector_type(8)))  float  v8f;

union Frag16 { unsigned int u[8]; v16bf v; };

// ---------------------------------------------------------------------------
// Generic batched WMMA GEMM (double-buffered LDS, register-staged prefetch):
//   C[z][m][n] = act( alpha * sum_k A[z][m][k] * B(z,k,n) + bias[n] )
//   TB==0 : B(z,k,n) = B[k*ldb + n]      (row-major K x N)
//   TB==1 : B(z,k,n) = B[n*ldb + k]      (row-major N x K, i.e. B transposed)
//   NF    : N is a multiple of 128 (no column bounds checks anywhere)
// Requirements (all call sites honor them): M % 128 == 0, K % 32 == 0, K >= 64,
// and for TB==1 additionally N % 128 == 0.
// Per-z offsets: z -> (b = z/nh, h = z%nh), ptr += b*s?b + h*s?h
// Tile: 128x128, 256 threads (8 waves), wave tile 32x64, K-step 32 (bf16).
// ---------------------------------------------------------------------------
#define BM 128
#define BN 128
#define BK 32
#define LDSP 36   // bf16 row stride (even -> dword-aligned pair reads)

template <int TB, int ACT, bool NF>
__global__ __launch_bounds__(256) void wmma_gemm(
    const float* __restrict__ A, const float* __restrict__ Bm,
    const float* __restrict__ bias, float* __restrict__ C,
    int M, int N, int K, int lda, int ldb, int ldc,
    long long sAb, long long sAh, long long sBb, long long sBh,
    long long sCb, long long sCh, int nh, float alpha)
{
    __shared__ __bf16 As[2][BM][LDSP];
    __shared__ __bf16 Bs[2][BN][LDSP];

    const int z  = blockIdx.z;
    const int bb = z / nh, hh = z % nh;
    A  += (long long)bb * sAb + (long long)hh * sAh;
    Bm += (long long)bb * sBb + (long long)hh * sBh;
    C  += (long long)bb * sCb + (long long)hh * sCh;

    const int m0 = blockIdx.y * BM;
    const int n0 = blockIdx.x * BN;
    const int tid  = threadIdx.x;
    const int lane = tid & 31;
    const int wid  = tid >> 5;      // 0..7
    const int wm   = wid & 3;       // 32-row group
    const int wn   = wid >> 2;      // 64-col group

    v8f acc[2][4];
    for (int i = 0; i < 2; ++i)
        for (int j = 0; j < 4; ++j)
            for (int e = 0; e < 8; ++e) acc[i][j][e] = 0.0f;

    const int mload = tid >> 3;         // 0..31
    const int kload = (tid & 7) * 4;    // 0,4,...,28
    const int krowB = tid >> 5;         // 0..7   (TB==0)
    const int ncolB = (tid & 31) * 4;   // 0..124 (TB==0)

    const int fl   = lane & 15;         // fragment row/col within 16
    const int fsel = lane >> 4;         // K-half selector

    float4 aR[4], bR[4];

    // stage one K-tile of A and B into registers (unguarded where legal)
    auto load_tiles = [&](int kt) {
#pragma unroll
        for (int p = 0; p < 4; ++p)
            aR[p] = *(const float4*)(A + (long long)(m0 + mload + p * 32) * lda + kt + kload);
        if (TB == 0) {
#pragma unroll
            for (int p = 0; p < 4; ++p) {
                int kr = krowB + p * 8;
                int gn = n0 + ncolB;
                if (NF) {
                    bR[p] = *(const float4*)(Bm + (long long)(kt + kr) * ldb + gn);
                } else {
                    float t0 = (gn + 0 < N) ? Bm[(long long)(kt + kr) * ldb + gn + 0] : 0.f;
                    float t1 = (gn + 1 < N) ? Bm[(long long)(kt + kr) * ldb + gn + 1] : 0.f;
                    float t2 = (gn + 2 < N) ? Bm[(long long)(kt + kr) * ldb + gn + 2] : 0.f;
                    float t3 = (gn + 3 < N) ? Bm[(long long)(kt + kr) * ldb + gn + 3] : 0.f;
                    bR[p] = make_float4(t0, t1, t2, t3);
                }
            }
        } else {
#pragma unroll
            for (int p = 0; p < 4; ++p)
                bR[p] = *(const float4*)(Bm + (long long)(n0 + mload + p * 32) * ldb + kt + kload);
        }
    };

    // convert staged registers to bf16 and commit to LDS buffer `bf_`
    auto store_tiles = [&](int bf_) {
#pragma unroll
        for (int p = 0; p < 4; ++p) {
            int row = mload + p * 32;
            As[bf_][row][kload + 0] = (__bf16)aR[p].x;
            As[bf_][row][kload + 1] = (__bf16)aR[p].y;
            As[bf_][row][kload + 2] = (__bf16)aR[p].z;
            As[bf_][row][kload + 3] = (__bf16)aR[p].w;
        }
        if (TB == 0) {
#pragma unroll
            for (int p = 0; p < 4; ++p) {
                int kr = krowB + p * 8;
                Bs[bf_][ncolB + 0][kr] = (__bf16)bR[p].x;
                Bs[bf_][ncolB + 1][kr] = (__bf16)bR[p].y;
                Bs[bf_][ncolB + 2][kr] = (__bf16)bR[p].z;
                Bs[bf_][ncolB + 3][kr] = (__bf16)bR[p].w;
            }
        } else {
#pragma unroll
            for (int p = 0; p < 4; ++p) {
                int nr = mload + p * 32;
                Bs[bf_][nr][kload + 0] = (__bf16)bR[p].x;
                Bs[bf_][nr][kload + 1] = (__bf16)bR[p].y;
                Bs[bf_][nr][kload + 2] = (__bf16)bR[p].z;
                Bs[bf_][nr][kload + 3] = (__bf16)bR[p].w;
            }
        }
    };

    const int nk = K / BK;

    load_tiles(0);
    store_tiles(0);

    for (int ki = 0; ki < nk; ++ki) {
        __syncthreads();                         // LDS[cur] ready for all waves
        const int cur = ki & 1;

        if (ki + 1 < nk) load_tiles((ki + 1) * BK);      // overlap with WMMA below
        if (ki + 2 < nk)                                  // L2 prefetch 2 tiles ahead
            __builtin_prefetch(A + (long long)(m0 + mload) * lda + (ki + 2) * BK + kload, 0, 1);

        // ---- fragments (ISA 16-bit A 16x32 / B 32x16 lane layouts)
        Frag16 af[2];
#pragma unroll
        for (int i = 0; i < 2; ++i) {
            int row = wm * 32 + i * 16 + fl;
#pragma unroll
            for (int j = 0; j < 8; ++j) {
                int k = (j < 4) ? (fsel * 8 + 2 * j) : (16 + fsel * 8 + 2 * (j - 4));
                af[i].u[j] = *(const unsigned int*)&As[cur][row][k];
            }
        }
        Frag16 bf[4];
#pragma unroll
        for (int j4 = 0; j4 < 4; ++j4) {
            int col = wn * 64 + j4 * 16 + fl;
#pragma unroll
            for (int j = 0; j < 8; ++j) {
                int k = (j < 4) ? (fsel * 8 + 2 * j) : (16 + fsel * 8 + 2 * (j - 4));
                bf[j4].u[j] = *(const unsigned int*)&Bs[cur][col][k];
            }
        }
#pragma unroll
        for (int i = 0; i < 2; ++i)
#pragma unroll
            for (int j = 0; j < 4; ++j)
                acc[i][j] = __builtin_amdgcn_wmma_f32_16x16x32_bf16(
                    false, af[i].v, false, bf[j].v, (short)0, acc[i][j], false, false);

        if (ki + 1 < nk) store_tiles(cur ^ 1);   // commit next tile to other buffer
    }

    // ---- epilogue: scale, bias, activation, store (rows always in range)
    for (int i = 0; i < 2; ++i) {
        for (int j = 0; j < 4; ++j) {
            int col = n0 + wn * 64 + j * 16 + fl;
            if (!NF && col >= N) continue;
            float bv = bias ? bias[col] : 0.0f;
#pragma unroll
            for (int v = 0; v < 8; ++v) {
                int row = m0 + wm * 32 + i * 16 + fsel * 8 + v;
                float val = acc[i][j][v] * alpha + bv;
                if (ACT == 1) val = fmaxf(val, 0.0f);
                else if (ACT == 2) val = 0.5f * val * (1.0f + erff(val * 0.70710678118654752f));
                C[(long long)row * ldc + col] = val;
            }
        }
    }
}

// ---------------------------------------------------------------------------
// Utility / elementwise kernels
// ---------------------------------------------------------------------------
__global__ void zero_f(float* p, long long n) {
    long long i = (long long)blockIdx.x * blockDim.x + threadIdx.x;
    if (i < n) p[i] = 0.0f;
}
__global__ void zero_u(unsigned int* p, long long n) {
    long long i = (long long)blockIdx.x * blockDim.x + threadIdx.x;
    if (i < n) p[i] = 0u;
}

// spatial embedding gather into cat1[:, 384:768]
__global__ void sp_embed_kernel(const int* __restrict__ bbox, const float* __restrict__ Ex,
                                const float* __restrict__ Ey, float* __restrict__ cat1) {
    int idx = blockIdx.x * blockDim.x + threadIdx.x;
    if (idx >= 4096 * 384) return;
    int n = idx / 384, c = idx % 384;
    int part = c / 96, cc = c % 96;
    const float* T = (part == 0 || part == 2) ? Ex : Ey;
    int bi = bbox[n * 4 + part];
    cat1[(long long)n * 1536 + 384 + c] = T[bi * 96 + cc];
}

__global__ void copyx_kernel(const float* __restrict__ x, float* __restrict__ cat1) {
    int idx = blockIdx.x * blockDim.x + threadIdx.x;
    if (idx >= 4096 * 768) return;
    int n = idx / 768, c = idx % 768;
    cat1[(long long)n * 1536 + 768 + c] = x[idx];
}

// ---------------------------------------------------------------------------
// GAT kernels
// ---------------------------------------------------------------------------
__global__ void deg_kernel(const int* __restrict__ ei, const float* __restrict__ ea,
                           float* cnt, float* sum, int E) {
    int e = blockIdx.x * blockDim.x + threadIdx.x;
    if (e >= E) return;
    int d = ei[E + e];
    atomicAdd(&cnt[d], 1.0f);
    atomicAdd(&sum[d], ea[e]);
}
__global__ void loopea_fin_kernel(float* sum, const float* cnt, int n) {
    int i = blockIdx.x * blockDim.x + threadIdx.x;
    if (i < n) sum[i] /= fmaxf(cnt[i], 1.0f);
}

__global__ void gat_coef_kernel(const float* __restrict__ h, const float* __restrict__ as_,
                                const float* __restrict__ ad_, float* asrc, float* adst) {
    int idx = blockIdx.x * blockDim.x + threadIdx.x;   // n*8+g
    if (idx >= 4096 * 8) return;
    int n = idx >> 3, g = idx & 7;
    const float* hp = h + (long long)n * 768 + g * 96;
    const float* ap = as_ + g * 96;
    const float* dp = ad_ + g * 96;
    float s = 0.f, d = 0.f;
    for (int c = 0; c < 96; ++c) { float hv = hp[c]; s += hv * ap[c]; d += hv * dp[c]; }
    asrc[idx] = s; adst[idx] = d;
}

__global__ void wdot_kernel(const float* __restrict__ We, const float* __restrict__ ae,
                            float* wdot) {
    int g = threadIdx.x;
    if (g >= 8) return;
    float s = 0.f;
    for (int c = 0; c < 96; ++c) s += We[g * 96 + c] * ae[g * 96 + c];
    wdot[g] = s;
}

__device__ __forceinline__ unsigned int fmapmax(float f) {
    unsigned int u = __float_as_uint(f);
    return (u & 0x80000000u) ? ~u : (u | 0x80000000u);
}
__device__ __forceinline__ float funmapmax(unsigned int u) {
    u = (u & 0x80000000u) ? (u & 0x7FFFFFFFu) : ~u;
    return __uint_as_float(u);
}

__global__ void gat_alpha_kernel(const int* __restrict__ ei, const float* __restrict__ ea,
                                 const float* __restrict__ loopea,
                                 const float* __restrict__ asrc, const float* __restrict__ adst,
                                 const float* __restrict__ wdot,
                                 float* alpha, unsigned int* mseg, int E, int E2) {
    int idx = blockIdx.x * blockDim.x + threadIdx.x;   // e*8+g
    if (idx >= E2 * 8) return;
    int e = idx >> 3, g = idx & 7;
    int src, dst; float eav;
    if (e < E) { src = ei[e]; dst = ei[E + e]; eav = ea[e]; }
    else       { src = dst = e - E; eav = loopea[src]; }
    float a = asrc[src * 8 + g] + adst[dst * 8 + g] + eav * wdot[g];
    a = (a > 0.f) ? a : 0.2f * a;                      // leaky relu, slope 0.2
    alpha[idx] = a;
    atomicMax(&mseg[dst * 8 + g], fmapmax(a));
}

__global__ void gat_exp_kernel(const int* __restrict__ ei, float* alpha,
                               const unsigned int* __restrict__ mseg, float* den,
                               int E, int E2) {
    int idx = blockIdx.x * blockDim.x + threadIdx.x;
    if (idx >= E2 * 8) return;
    int e = idx >> 3, g = idx & 7;
    int dst = (e < E) ? ei[E + e] : (e - E);
    float m  = funmapmax(mseg[dst * 8 + g]);
    float ex = expf(alpha[idx] - m);
    alpha[idx] = ex;
    atomicAdd(&den[dst * 8 + g], ex);
}

__global__ void gat_msg_kernel(const int* __restrict__ ei, const float* __restrict__ alpha,
                               const float* __restrict__ den, const float* __restrict__ h,
                               float* out, int E, int E2) {
    int e = blockIdx.x;
    int src, dst;
    if (e < E) { src = ei[e]; dst = ei[E + e]; }
    else       { src = dst = e - E; }
    __shared__ float w[8];
    int tid = threadIdx.x;
    if (tid < 8) w[tid] = alpha[e * 8 + tid] / (den[dst * 8 + tid] + 1e-16f);
    __syncthreads();
    for (int c = tid; c < 768; c += 256)
        atomicAdd(&out[(long long)dst * 768 + c], w[c / 96] * h[(long long)src * 768 + c]);
}

// x = LN(relu(acc + b0 + b1 + b2)) * g + b, eps 1e-5
__global__ void gat_post_kernel(const float* __restrict__ acc,
                                const float* __restrict__ b0, const float* __restrict__ b1,
                                const float* __restrict__ b2,
                                const float* __restrict__ g, const float* __restrict__ b,
                                float* x) {
    int n = blockIdx.x, tid = threadIdx.x;
    __shared__ float red[256];
    float v[3], s = 0.f;
    for (int i = 0; i < 3; ++i) {
        int c = tid + i * 256;
        float t = acc[(long long)n * 768 + c] + b0[c] + b1[c] + b2[c];
        t = fmaxf(t, 0.f);
        v[i] = t; s += t;
    }
    red[tid] = s; __syncthreads();
    for (int o = 128; o > 0; o >>= 1) { if (tid < o) red[tid] += red[tid + o]; __syncthreads(); }
    float mu = red[0] / 768.f; __syncthreads();
    float s2 = 0.f;
    for (int i = 0; i < 3; ++i) { float d = v[i] - mu; s2 += d * d; }
    red[tid] = s2; __syncthreads();
    for (int o = 128; o > 0; o >>= 1) { if (tid < o) red[tid] += red[tid + o]; __syncthreads(); }
    float inv = rsqrtf(red[0] / 768.f + 1e-5f);
    for (int i = 0; i < 3; ++i) {
        int c = tid + i * 256;
        x[(long long)n * 768 + c] = (v[i] - mu) * inv * g[c] + b[c];
    }
}

// h = LN(h + pos[s] + type), eps 1e-12 (in-place)
__global__ void embed_ln_kernel(float* h, const float* __restrict__ pos,
                                const float* __restrict__ typ,
                                const float* __restrict__ g, const float* __restrict__ b) {
    int n = blockIdx.x, tid = threadIdx.x;
    int s = n % 512;
    __shared__ float red[256];
    float v[3], sm = 0.f;
    for (int i = 0; i < 3; ++i) {
        int c = tid + i * 256;
        float t = h[(long long)n * 768 + c] + pos[(long long)s * 768 + c] + typ[c];
        v[i] = t; sm += t;
    }
    red[tid] = sm; __syncthreads();
    for (int o = 128; o > 0; o >>= 1) { if (tid < o) red[tid] += red[tid + o]; __syncthreads(); }
    float mu = red[0] / 768.f; __syncthreads();
    float s2 = 0.f;
    for (int i = 0; i < 3; ++i) { float d = v[i] - mu; s2 += d * d; }
    red[tid] = s2; __syncthreads();
    for (int o = 128; o > 0; o >>= 1) { if (tid < o) red[tid] += red[tid + o]; __syncthreads(); }
    float inv = rsqrtf(red[0] / 768.f + 1e-12f);
    for (int i = 0; i < 3; ++i) {
        int c = tid + i * 256;
        h[(long long)n * 768 + c] = (v[i] - mu) * inv * g[c] + b[c];
    }
}

// h = LN(h + t), eps 1e-12 (in-place)
__global__ void res_ln_kernel(float* h, const float* __restrict__ t,
                              const float* __restrict__ g, const float* __restrict__ b) {
    int n = blockIdx.x, tid = threadIdx.x;
    __shared__ float red[256];
    float v[3], sm = 0.f;
    for (int i = 0; i < 3; ++i) {
        int c = tid + i * 256;
        float u = h[(long long)n * 768 + c] + t[(long long)n * 768 + c];
        v[i] = u; sm += u;
    }
    red[tid] = sm; __syncthreads();
    for (int o = 128; o > 0; o >>= 1) { if (tid < o) red[tid] += red[tid + o]; __syncthreads(); }
    float mu = red[0] / 768.f; __syncthreads();
    float s2 = 0.f;
    for (int i = 0; i < 3; ++i) { float d = v[i] - mu; s2 += d * d; }
    red[tid] = s2; __syncthreads();
    for (int o = 128; o > 0; o >>= 1) { if (tid < o) red[tid] += red[tid + o]; __syncthreads(); }
    float inv = rsqrtf(red[0] / 768.f + 1e-12f);
    for (int i = 0; i < 3; ++i) {
        int c = tid + i * 256;
        h[(long long)n * 768 + c] = (v[i] - mu) * inv * g[c] + b[c];
    }
}

// row softmax over 512 cols (scores already scaled by 1/8 in GEMM epilogue)
__global__ void softmax_kernel(float* sc) {
    long long r = blockIdx.x;
    float* p = sc + r * 512;
    int tid = threadIdx.x;
    __shared__ float red[256];
    float a0 = p[tid], a1 = p[tid + 256];
    red[tid] = fmaxf(a0, a1); __syncthreads();
    for (int o = 128; o > 0; o >>= 1) { if (tid < o) red[tid] = fmaxf(red[tid], red[tid + o]); __syncthreads(); }
    float mx = red[0]; __syncthreads();
    float e0 = expf(a0 - mx), e1 = expf(a1 - mx);
    red[tid] = e0 + e1; __syncthreads();
    for (int o = 128; o > 0; o >>= 1) { if (tid < o) red[tid] += red[tid + o]; __syncthreads(); }
    float inv = 1.0f / red[0];
    p[tid] = e0 * inv;
    p[tid + 256] = e1 * inv;
}

// out[n, 0..4] = h[n,:] @ Wout + bout  (one wave per output column)
__global__ void out_proj_kernel(const float* __restrict__ h, const float* __restrict__ W,
                                const float* __restrict__ b, float* out) {
    int n = blockIdx.x;
    int wid = threadIdx.x >> 5, lane = threadIdx.x & 31;
    if (wid >= 5) return;
    float s = 0.f;
    for (int c = lane; c < 768; c += 32) s += h[(long long)n * 768 + c] * W[c * 5 + wid];
    for (int o = 16; o > 0; o >>= 1) s += __shfl_down(s, o, 32);
    if (lane == 0) out[n * 5 + wid] = s + b[wid];
}

// ---------------------------------------------------------------------------
// Host-side GEMM dispatch
// ---------------------------------------------------------------------------
static void launch_gemm(hipStream_t s, int TB, int ACT,
                        const float* A, const float* B, const float* bias, float* C,
                        int M, int N, int K, int lda, int ldb, int ldc,
                        long long sAb, long long sAh, long long sBb, long long sBh,
                        long long sCb, long long sCh, int nh, int nz, float alpha) {
    dim3 grd((N + BN - 1) / BN, (M + BM - 1) / BM, nz), blk(256);
    const bool nf = (N % BN) == 0;
    if (TB == 1)
        wmma_gemm<1, 0, true><<<grd, blk, 0, s>>>(A, B, bias, C, M, N, K, lda, ldb, ldc,
                                                  sAb, sAh, sBb, sBh, sCb, sCh, nh, alpha);
    else if (ACT == 1)
        wmma_gemm<0, 1, true><<<grd, blk, 0, s>>>(A, B, bias, C, M, N, K, lda, ldb, ldc,
                                                  sAb, sAh, sBb, sBh, sCb, sCh, nh, alpha);
    else if (ACT == 2)
        wmma_gemm<0, 2, true><<<grd, blk, 0, s>>>(A, B, bias, C, M, N, K, lda, ldb, ldc,
                                                  sAb, sAh, sBb, sBh, sCb, sCh, nh, alpha);
    else if (nf)
        wmma_gemm<0, 0, true><<<grd, blk, 0, s>>>(A, B, bias, C, M, N, K, lda, ldb, ldc,
                                                  sAb, sAh, sBb, sBh, sCb, sCh, nh, alpha);
    else
        wmma_gemm<0, 0, false><<<grd, blk, 0, s>>>(A, B, bias, C, M, N, K, lda, ldb, ldc,
                                                   sAb, sAh, sBb, sBh, sCb, sCh, nh, alpha);
}

static inline int cdiv(long long a, int b) { return (int)((a + b - 1) / b); }

// ---------------------------------------------------------------------------
// kernel_launch
// ---------------------------------------------------------------------------
extern "C" void kernel_launch(void* const* d_in, const int* in_sizes, int n_in,
                              void* d_out, int out_size, void* d_ws, size_t ws_size,
                              hipStream_t stream) {
    (void)n_in; (void)out_size; (void)ws_size;

    const float* text_x = (const float*)d_in[0];
    const float* img_x  = (const float*)d_in[1];
    const int*   bbox   = (const int*)d_in[2];
    const int*   ei[3]  = {(const int*)d_in[3], (const int*)d_in[4], (const int*)d_in[5]};
    const float* ea[3]  = {(const float*)d_in[6], (const float*)d_in[7], (const float*)d_in[8]};
    const float* Wt = (const float*)d_in[9],  *bt = (const float*)d_in[10];
    const float* Wi = (const float*)d_in[11], *bi = (const float*)d_in[12];
    const float* Ex = (const float*)d_in[13], *Ey = (const float*)d_in[14];
    const float* Wc = (const float*)d_in[15], *bc = (const float*)d_in[16];
    const float* gW  = (const float*)d_in[17];
    const float* gas = (const float*)d_in[18], *gad = (const float*)d_in[19];
    const float* gWe = (const float*)d_in[20], *gae = (const float*)d_in[21];
    const float* gb  = (const float*)d_in[22];
    const float* lng = (const float*)d_in[23], *lnb = (const float*)d_in[24];
    const float* Wcomb = (const float*)d_in[25], *bcomb = (const float*)d_in[26];
    const float* pos = (const float*)d_in[27], *typ = (const float*)d_in[28];
    const float* eg  = (const float*)d_in[29], *eb  = (const float*)d_in[30];
    const float* Wq = (const float*)d_in[31], *Wk = (const float*)d_in[32];
    const float* Wv = (const float*)d_in[33], *Wo = (const float*)d_in[34];
    const float* W1 = (const float*)d_in[35], *W2 = (const float*)d_in[36];
    const float* bq = (const float*)d_in[37], *bk = (const float*)d_in[38];
    const float* bv = (const float*)d_in[39], *bo = (const float*)d_in[40];
    const float* b1 = (const float*)d_in[41], *b2 = (const float*)d_in[42];
    const float* l1g = (const float*)d_in[43], *l1b = (const float*)d_in[44];
    const float* l2g = (const float*)d_in[45], *l2b = (const float*)d_in[46];
    const float* WoutP = (const float*)d_in[47], *boutP = (const float*)d_in[48];

    const int NN = 4096, S = 512, BH = 768, NH = 12, HD = 64, FF = 3072, LB = 12;
    const int E = in_sizes[3] / 2;          // 65536
    const int E2 = E + NN;

    // ---- workspace carve-up
    char* cur = (char*)d_ws;
    auto alloc = [&](size_t nfl) -> float* {
        float* p = (float*)cur;
        cur += ((nfl * sizeof(float)) + 255) / 256 * 256;
        return p;
    };
    float* cat1  = alloc((size_t)NN * 1536);
    float* xb    = alloc((size_t)NN * BH);
    float* gath  = alloc((size_t)NN * BH);
    float* gout  = alloc((size_t)NN * BH);
    float* asrc  = alloc((size_t)NN * 8);
    float* adst  = alloc((size_t)NN * 8);
    float* cnt   = alloc(NN);
    float* lea[3]; for (int t = 0; t < 3; ++t) lea[t] = alloc(NN);
    float* wdt   = alloc(8);
    float* albuf = alloc((size_t)E2 * 8);
    unsigned int* mseg = (unsigned int*)alloc((size_t)NN * 8);
    float* den   = alloc((size_t)NN * 8);
    float* hseq  = alloc((size_t)NN * BH);
    float* qb2   = alloc((size_t)NN * BH);
    float* kb2   = alloc((size_t)NN * BH);
    float* vb2   = alloc((size_t)NN * BH);
    float* ctxb  = alloc((size_t)NN * BH);
    float* tmpb  = alloc((size_t)NN * BH);
    float* ffb   = alloc((size_t)NN * FF);
    float* scb   = alloc((size_t)8 * NH * S * S);

    // ---- 1. node projections into cat1 = [tp | ip | sp | (x later)]
    launch_gemm(stream, 0, 0, text_x, Wt, bt, cat1 + 0,   NN, 192, 768, 768, 192, 1536, 0,0,0,0,0,0, 1, 1, 1.f);
    launch_gemm(stream, 0, 0, img_x,  Wi, bi, cat1 + 192, NN, 192, 512, 512, 192, 1536, 0,0,0,0,0,0, 1, 1, 1.f);
    sp_embed_kernel<<<cdiv((long long)NN * 384, 256), 256, 0, stream>>>(bbox, Ex, Ey, cat1);
    // x = relu(cat0 @ Wc + bc)
    launch_gemm(stream, 0, 1, cat1, Wc, bc, xb, NN, BH, 768, 1536, BH, BH, 0,0,0,0,0,0, 1, 1, 1.f);

    // ---- 2. per-type self-loop edge-attr means (layer invariant)
    for (int t = 0; t < 3; ++t) {
        zero_f<<<cdiv(NN, 256), 256, 0, stream>>>(cnt, NN);
        zero_f<<<cdiv(NN, 256), 256, 0, stream>>>(lea[t], NN);
        deg_kernel<<<cdiv(E, 256), 256, 0, stream>>>(ei[t], ea[t], cnt, lea[t], E);
        loopea_fin_kernel<<<cdiv(NN, 256), 256, 0, stream>>>(lea[t], cnt, NN);
    }

    // ---- 3. GAT layers
    for (int l = 0; l < 2; ++l) {
        zero_f<<<cdiv((long long)NN * BH, 256), 256, 0, stream>>>(gout, (long long)NN * BH);
        for (int t = 0; t < 3; ++t) {
            size_t o = (size_t)(l * 3 + t);
            launch_gemm(stream, 0, 0, xb, gW + o * 768 * 768, nullptr, gath,
                        NN, BH, BH, BH, BH, BH, 0,0,0,0,0,0, 1, 1, 1.f);
            gat_coef_kernel<<<cdiv((long long)NN * 8, 256), 256, 0, stream>>>(
                gath, gas + o * 768, gad + o * 768, asrc, adst);
            wdot_kernel<<<1, 32, 0, stream>>>(gWe + o * 768, gae + o * 768, wdt);
            zero_u<<<cdiv((long long)NN * 8, 256), 256, 0, stream>>>(mseg, (long long)NN * 8);
            zero_f<<<cdiv((long long)NN * 8, 256), 256, 0, stream>>>(den, (long long)NN * 8);
            gat_alpha_kernel<<<cdiv((long long)E2 * 8, 256), 256, 0, stream>>>(
                ei[t], ea[t], lea[t], asrc, adst, wdt, albuf, mseg, E, E2);
            gat_exp_kernel<<<cdiv((long long)E2 * 8, 256), 256, 0, stream>>>(
                ei[t], albuf, mseg, den, E, E2);
            gat_msg_kernel<<<E2, 256, 0, stream>>>(ei[t], albuf, den, gath, gout, E, E2);
        }
        gat_post_kernel<<<NN, 256, 0, stream>>>(gout,
            gb + (size_t)(l * 3 + 0) * 768, gb + (size_t)(l * 3 + 1) * 768,
            gb + (size_t)(l * 3 + 2) * 768, lng + (size_t)l * 768, lnb + (size_t)l * 768, xb);
    }

    // ---- 4. combine + embeddings
    copyx_kernel<<<cdiv((long long)NN * BH, 256), 256, 0, stream>>>(xb, cat1);
    launch_gemm(stream, 0, 1, cat1, Wcomb, bcomb, hseq, NN, BH, 1536, 1536, BH, BH,
                0,0,0,0,0,0, 1, 1, 1.f);
    embed_ln_kernel<<<NN, 256, 0, stream>>>(hseq, pos, typ, eg, eb);

    // ---- 5. transformer encoder (12 layers)
    for (int l = 0; l < LB; ++l) {
        const size_t wo = (size_t)l * BH * BH;
        launch_gemm(stream, 0, 0, hseq, Wq + wo, bq + (size_t)l * BH, qb2,
                    NN, BH, BH, BH, BH, BH, 0,0,0,0,0,0, 1, 1, 1.f);
        launch_gemm(stream, 0, 0, hseq, Wk + wo, bk + (size_t)l * BH, kb2,
                    NN, BH, BH, BH, BH, BH, 0,0,0,0,0,0, 1, 1, 1.f);
        launch_gemm(stream, 0, 0, hseq, Wv + wo, bv + (size_t)l * BH, vb2,
                    NN, BH, BH, BH, BH, BH, 0,0,0,0,0,0, 1, 1, 1.f);
        // scores[b,h] = (Q K^T) / 8  : batched over z = b*NH + h
        launch_gemm(stream, 1, 0, qb2, kb2, nullptr, scb,
                    S, S, HD, BH, BH, S,
                    (long long)S * BH, HD, (long long)S * BH, HD,
                    (long long)NH * S * S, (long long)S * S,
                    NH, 8 * NH, 0.125f);
        softmax_kernel<<<8 * NH * S, 256, 0, stream>>>(scb);
        // ctx[b,h] = P V
        launch_gemm(stream, 0, 0, scb, vb2, nullptr, ctxb,
                    S, HD, S, S, BH, BH,
                    (long long)NH * S * S, (long long)S * S,
                    (long long)S * BH, HD, (long long)S * BH, HD,
                    NH, 8 * NH, 1.f);
        launch_gemm(stream, 0, 0, ctxb, Wo + wo, bo + (size_t)l * BH, tmpb,
                    NN, BH, BH, BH, BH, BH, 0,0,0,0,0,0, 1, 1, 1.f);
        res_ln_kernel<<<NN, 256, 0, stream>>>(hseq, tmpb, l1g + (size_t)l * BH, l1b + (size_t)l * BH);
        launch_gemm(stream, 0, 2, hseq, W1 + (size_t)l * BH * FF, b1 + (size_t)l * FF, ffb,
                    NN, FF, BH, BH, FF, FF, 0,0,0,0,0,0, 1, 1, 1.f);
        launch_gemm(stream, 0, 0, ffb, W2 + (size_t)l * FF * BH, b2 + (size_t)l * BH, tmpb,
                    NN, BH, FF, FF, BH, BH, 0,0,0,0,0,0, 1, 1, 1.f);
        res_ln_kernel<<<NN, 256, 0, stream>>>(hseq, tmpb, l2g + (size_t)l * BH, l2b + (size_t)l * BH);
    }

    // ---- 6. output head
    out_proj_kernel<<<NN, 256, 0, stream>>>(hseq, WoutP, boutP, (float*)d_out);
}
